// SphericalConv2d_24635932410144
// MI455X (gfx1250) — matrix operand order repaired and checked by reading, not compiled, not run on hardware
//
#include <hip/hip_runtime.h>
#include <math.h>

// SphericalConv2d on MW grid as 5 stages (4 WMMA GEMMs + VALU channel mix) (gfx1250, wave32).
// f16 operands, f32 accumulation via v_wmma_f32_16x16x32_f16.
// B matrices are precomputed tables stored in WMMA-fragment-swizzled order so a
// lane's B fragment is one contiguous 32B load. Per-m Legendre stages stage the
// 32KB B block in LDS via global_load_async_to_lds_b128 (ASYNCcnt) so B reads
// use the DS pipe while A streams through VMEM.

#define LBAND 128
#define TDIM  128
#define NPHI  255
#define NPAD  256
#define MDIM  128
#define BATCH_N 16
#define CIN   8
#define COUT  8

typedef __attribute__((ext_vector_type(16))) _Float16 v16h;
typedef __attribute__((ext_vector_type(8)))  _Float16 v8h;
typedef __attribute__((ext_vector_type(8)))  float    v8f;

// Swizzled B element address for logical B[k][n] with K rows:
// lane(col=n&15, half) reads halves (half*16+e), e=0..15, contiguously per k-step.
__device__ __forceinline__ size_t bswz(int k, int n, int K) {
  return ((((size_t)(n >> 4) * (K >> 5) + (k >> 5)) * 16 + (n & 15)) << 5) + (k & 31);
}

// ---------------------------------------------------------------------------
// 32x32 macro-tile (2x2 WMMA tiles) : C[32x32] = A[32xK] * B[Kx32]
// A row-major f16 (lda halves); B0/B1 = swizzled 16-wide column-tile bases
// (column tile c base = Bs + c*K*16). acc = {c00,c01,c10,c11}.
// Works with A/B in global or LDS (addrspace inferred after inlining).
// ---------------------------------------------------------------------------
template<int K>
__device__ __forceinline__ void mma32x32(const _Float16* A, int lda,
                                         const _Float16* B0,
                                         const _Float16* B1,
                                         v8f acc[4]) {
  const int lane = threadIdx.x & 31;
  const int row  = lane & 15;          // also the B column within the tile
  const int half = lane >> 4;
  #pragma unroll
  for (int k0 = 0; k0 < K; k0 += 32) {
    const _Float16* ap0 = A + (size_t)row * lda + k0 + half * 16;
    const _Float16* ap1 = ap0 + (size_t)16 * lda;
    v8h a0l = *(const v8h*)ap0, a0h = *(const v8h*)(ap0 + 8);
    v8h a1l = *(const v8h*)ap1, a1h = *(const v8h*)(ap1 + 8);
    v16h a0 = __builtin_shufflevector(a0l, a0h, 0,1,2,3,4,5,6,7,8,9,10,11,12,13,14,15);
    v16h a1 = __builtin_shufflevector(a1l, a1h, 0,1,2,3,4,5,6,7,8,9,10,11,12,13,14,15);
    size_t boff = ((((size_t)(k0 >> 5) * 16) + row) << 5) + half * 16;
    v16h b0 = *(const v16h*)(B0 + boff);   // 32B aligned contiguous fragment
    v16h b1 = *(const v16h*)(B1 + boff);
    acc[0] = __builtin_amdgcn_wmma_f32_16x16x32_f16(false, a0, false, b0, (short)0, acc[0], false, false);
    acc[1] = __builtin_amdgcn_wmma_f32_16x16x32_f16(false, a0, false, b1, (short)0, acc[1], false, false);
    acc[2] = __builtin_amdgcn_wmma_f32_16x16x32_f16(false, a1, false, b0, (short)0, acc[2], false, false);
    acc[3] = __builtin_amdgcn_wmma_f32_16x16x32_f16(false, a1, false, b1, (short)0, acc[3], false, false);
  }
}

// Async copy of one 16B chunk global -> LDS (CDNA5 Tensor/async path, ASYNCcnt).
__device__ __forceinline__ void async_b128(const _Float16* g, _Float16* lds) {
  unsigned loff = (unsigned)(size_t)(void*)lds;          // low 32 bits = LDS offset
  unsigned long long ga = (unsigned long long)g;
  asm volatile("global_load_async_to_lds_b128 %0, %1, off"
               :: "v"(loff), "v"(ga) : "memory");
}
__device__ __forceinline__ void wait_async0() {
  asm volatile("s_wait_asynccnt 0" ::: "memory");
}

// ---------------------------------------------------------------------------
// Table builders / converts (write B matrices pre-swizzled)
// ---------------------------------------------------------------------------
__global__ void k_build_tables(_Float16* __restrict__ dftBs, _Float16* __restrict__ idftBs) {
  int idx = blockIdx.x * blockDim.x + threadIdx.x;   // 0..65535 over [256][256]
  int a = idx >> 8, c = idx & 255;
  const float TWO_PI = 6.283185307179586f;
  // Forward: logical dftB[k=p=a][n=c]; n<128 -> cos(2pi*n*p/255)/255 ; else -sin/255
  float v0 = 0.0f;
  if (a < NPHI) {
    int m = c & 127;
    float ang = TWO_PI * (float)m * (float)a / (float)NPHI;
    v0 = ((c < 128) ? cosf(ang) : -sinf(ang)) * (1.0f / (float)NPHI);
  }
  dftBs[bswz(a, c, NPAD)] = (_Float16)v0;
  // Inverse real half-spectrum: logical idftB[k=a][p=c]
  float v1 = 0.0f;
  if (c < NPHI) {
    int m = a & 127;
    float ang = TWO_PI * (float)m * (float)c / (float)NPHI;
    if (a < 128) v1 = ((m == 0) ? 1.0f : 2.0f) * cosf(ang);
    else         v1 = ((m == 0) ? 0.0f : -2.0f) * sinf(ang);
  }
  idftBs[bswz(a, c, NPAD)] = (_Float16)v1;
}

__global__ void k_cvt_pad(const float* __restrict__ in, _Float16* __restrict__ out, int rows) {
  int idx = blockIdx.x * blockDim.x + threadIdx.x;   // rows*256
  int r = idx >> 8, c = idx & 255;
  if (r >= rows) return;
  out[idx] = (c < NPHI) ? (_Float16)in[(size_t)r * NPHI + c] : (_Float16)0.0f;
}

// Pinv[j=m+127][l][t] -> PinvBs[m]{k=t, n=l}   (forward Legendre B)
// Psyn[j=m+127][t][l] -> PsynBs[m]{k=l, n=t}   (inverse Legendre B)
__global__ void k_cvt_pmats(const float* __restrict__ Psyn, const float* __restrict__ Pinv,
                            _Float16* __restrict__ PinvBs, _Float16* __restrict__ PsynBs) {
  int idx = blockIdx.x * blockDim.x + threadIdx.x;   // 128*128*128
  int l = idx & 127, t = (idx >> 7) & 127, m = idx >> 14;
  int j = m + (LBAND - 1);
  PinvBs[(size_t)m * TDIM * LBAND + bswz(t, l, TDIM)]  = (_Float16)Pinv[((size_t)j * LBAND + l) * TDIM + t];
  PsynBs[(size_t)m * LBAND * TDIM + bswz(l, t, LBAND)] = (_Float16)Psyn[((size_t)j * TDIM + t) * LBAND + l];
}

// ---------------------------------------------------------------------------
// Stage 1: forward DFT over phi. Ah[R][256] * dftBs -> Fout[plane][m][bi][t]
// ---------------------------------------------------------------------------
__global__ void k_dft_fwd(const _Float16* __restrict__ Ah, const _Float16* __restrict__ Bs,
                          _Float16* __restrict__ Fout, int RI) {
  int mt = blockIdx.x * (blockDim.x >> 5) + (threadIdx.x >> 5);
  int trm = mt >> 3, tcm = mt & 7;                 // 32-row block, 32-col block
  v8f acc[4] = {};
  const _Float16* B0 = Bs + (size_t)(tcm * 2) * NPAD * 16;
  mma32x32<NPAD>(Ah + (size_t)trm * 32 * NPAD, NPAD, B0, B0 + (size_t)NPAD * 16, acc);
  int lane = threadIdx.x & 31, half = lane >> 4, ccol = lane & 15;
  #pragma unroll
  for (int ti = 0; ti < 2; ++ti)
    #pragma unroll
    for (int tj = 0; tj < 2; ++tj) {
      int gc = tcm * 32 + tj * 16 + ccol;
      int plane = gc >> 7, m = gc & 127;
      #pragma unroll
      for (int v = 0; v < 8; ++v) {
        int gr = trm * 32 + ti * 16 + half * 8 + v;
        int bi = gr >> 7, t = gr & 127;
        Fout[(((size_t)plane * MDIM + m) * RI + bi) * TDIM + t] = (_Float16)acc[ti * 2 + tj][v];
      }
    }
}

// ---------------------------------------------------------------------------
// Stage 2: per-m forward Legendre. One block (8 waves) per (plane,m).
// B block (32KB) async-staged to LDS; A streams from global (VMEM+DS pipes).
// ---------------------------------------------------------------------------
__global__ void k_legendre_fwd(const _Float16* __restrict__ F, const _Float16* __restrict__ Ps,
                               _Float16* __restrict__ Xout, int RI) {
  __shared__ _Float16 sB[TDIM * LBAND];            // 32KB swizzled B block for this m
  int pm = blockIdx.x;                             // 0..255 = plane*128+m
  int plane = pm >> 7, m = pm & 127;
  const _Float16* gA = F  + (size_t)(plane * MDIM + m) * RI * TDIM;
  const _Float16* gB = Ps + (size_t)m * TDIM * LBAND;
  int tid = threadIdx.x;                           // 256 threads
  #pragma unroll
  for (int i = 0; i < 8; ++i) {                    // 2048 x 16B chunks
    int e = (tid + i * 256) * 8;
    async_b128(gB + e, &sB[e]);
  }
  wait_async0();
  __syncthreads();

  int wave = tid >> 5;
  int nmt = (RI >> 5) * 4;                         // 16 (x) or 8 (w) macro tiles
  _Float16* Obase = Xout + (size_t)(plane * MDIM + m) * RI * LBAND;
  int lane = tid & 31, half = lane >> 4, ccol = lane & 15;
  for (int mt = wave; mt < nmt; mt += 8) {
    int trm = mt >> 2, tcm = mt & 3;
    v8f acc[4] = {};
    const _Float16* B0 = sB + (size_t)(tcm * 2) * TDIM * 16;
    mma32x32<TDIM>(gA + (size_t)trm * 32 * TDIM, TDIM, B0, B0 + (size_t)TDIM * 16, acc);
    #pragma unroll
    for (int ti = 0; ti < 2; ++ti)
      #pragma unroll
      for (int tj = 0; tj < 2; ++tj) {
        int gc = tcm * 32 + tj * 16 + ccol;
        #pragma unroll
        for (int v = 0; v < 8; ++v) {
          int gr = trm * 32 + ti * 16 + half * 8 + v;
          Obase[(size_t)gr * LBAND + gc] = (_Float16)acc[ti * 2 + tj][v];
        }
      }
  }
}

// ---------------------------------------------------------------------------
// Stage 3: harmonic-domain channel contraction (complex, K=CIN=8) -> VALU.
// ---------------------------------------------------------------------------
__global__ void k_chanmix(const _Float16* __restrict__ X, const _Float16* __restrict__ W,
                          _Float16* __restrict__ O) {
  int idx = blockIdx.x * blockDim.x + threadIdx.x;   // 128*16*8*128
  int l = idx & 127, o = (idx >> 7) & 7, b = (idx >> 10) & 15, m = idx >> 14;
  const size_t xPlane = (size_t)MDIM * (BATCH_N * CIN) * LBAND;
  const size_t wPlane = (size_t)MDIM * (COUT * CIN) * LBAND;
  const size_t oPlane = (size_t)MDIM * (BATCH_N * COUT) * LBAND;
  float ore = 0.f, oim = 0.f;
  #pragma unroll
  for (int i = 0; i < CIN; ++i) {
    size_t xi = ((size_t)m * (BATCH_N * CIN) + b * CIN + i) * LBAND + l;
    size_t wi = ((size_t)m * (COUT * CIN) + o * CIN + i) * LBAND + l;
    float xr = (float)X[xi], xj = (float)X[xi + xPlane];
    float wr = (float)W[wi], wj = (float)W[wi + wPlane];
    ore += xr * wr - xj * wj;
    oim += xr * wj + xj * wr;
  }
  size_t oi = ((size_t)m * (BATCH_N * COUT) + b * COUT + o) * LBAND + l;
  O[oi]          = (_Float16)ore;
  O[oi + oPlane] = (_Float16)oim;
}

// ---------------------------------------------------------------------------
// Stage 4: per-m inverse Legendre -> G[(bo*128+t)][plane*128+m]
// Same LDS-staged structure as stage 2.
// ---------------------------------------------------------------------------
__global__ void k_legendre_inv(const _Float16* __restrict__ Olm, const _Float16* __restrict__ Ps,
                               _Float16* __restrict__ G) {
  __shared__ _Float16 sB[LBAND * TDIM];            // 32KB swizzled Psyn[m]
  const int RI = BATCH_N * COUT;                   // 128
  int pm = blockIdx.x;
  int plane = pm >> 7, m = pm & 127;
  const _Float16* gA = Olm + (size_t)(plane * MDIM + m) * RI * LBAND;
  const _Float16* gB = Ps  + (size_t)m * LBAND * TDIM;
  int tid = threadIdx.x;
  #pragma unroll
  for (int i = 0; i < 8; ++i) {
    int e = (tid + i * 256) * 8;
    async_b128(gB + e, &sB[e]);
  }
  wait_async0();
  __syncthreads();

  int wave = tid >> 5;
  int lane = tid & 31, half = lane >> 4, ccol = lane & 15;
  for (int mt = wave; mt < 16; mt += 8) {
    int trm = mt >> 2, tcm = mt & 3;
    v8f acc[4] = {};
    const _Float16* B0 = sB + (size_t)(tcm * 2) * LBAND * 16;
    mma32x32<LBAND>(gA + (size_t)trm * 32 * LBAND, LBAND, B0, B0 + (size_t)LBAND * 16, acc);
    #pragma unroll
    for (int ti = 0; ti < 2; ++ti)
      #pragma unroll
      for (int tj = 0; tj < 2; ++tj) {
        int t = tcm * 32 + tj * 16 + ccol;
        #pragma unroll
        for (int v = 0; v < 8; ++v) {
          int bo = trm * 32 + ti * 16 + half * 8 + v;
          G[((size_t)bo * TDIM + t) * NPAD + plane * MDIM + m] = (_Float16)acc[ti * 2 + tj][v];
        }
      }
  }
}

// ---------------------------------------------------------------------------
// Stage 5: inverse real DFT over phi + bias -> f32 output
// ---------------------------------------------------------------------------
__global__ void k_idft(const _Float16* __restrict__ G, const _Float16* __restrict__ Bs,
                       const float* __restrict__ bias, float* __restrict__ out) {
  int mt = blockIdx.x * (blockDim.x >> 5) + (threadIdx.x >> 5);
  int trm = mt >> 3, tcm = mt & 7;
  v8f acc[4] = {};
  const _Float16* B0 = Bs + (size_t)(tcm * 2) * NPAD * 16;
  mma32x32<NPAD>(G + (size_t)trm * 32 * NPAD, NPAD, B0, B0 + (size_t)NPAD * 16, acc);
  int lane = threadIdx.x & 31, half = lane >> 4, ccol = lane & 15;
  #pragma unroll
  for (int ti = 0; ti < 2; ++ti)
    #pragma unroll
    for (int tj = 0; tj < 2; ++tj) {
      int p = tcm * 32 + tj * 16 + ccol;
      if (p < NPHI) {
        #pragma unroll
        for (int v = 0; v < 8; ++v) {
          int r = trm * 32 + ti * 16 + half * 8 + v;   // r = (b*8+o)*128 + t
          int o = (r >> 7) & 7;
          out[(size_t)r * NPHI + p] = acc[ti * 2 + tj][v] + bias[o];
        }
      }
    }
}

// ---------------------------------------------------------------------------
extern "C" void kernel_launch(void* const* d_in, const int* in_sizes, int n_in,
                              void* d_out, int out_size, void* d_ws, size_t ws_size,
                              hipStream_t stream) {
  (void)in_sizes; (void)n_in; (void)out_size; (void)ws_size;
  const float* x    = (const float*)d_in[0];
  const float* wgt  = (const float*)d_in[1];
  const float* bias = (const float*)d_in[2];
  const float* Psyn = (const float*)d_in[3];
  const float* Pinv = (const float*)d_in[4];
  float* out = (float*)d_out;

  char* ws = (char*)d_ws;
  size_t off = 0;
  auto alloc = [&](size_t bytes) -> void* {
    void* p = ws + off; off += (bytes + 255) & ~(size_t)255; return p;
  };
  _Float16* xh     = (_Float16*)alloc((size_t)16384 * NPAD * 2);
  _Float16* wh     = (_Float16*)alloc((size_t)8192  * NPAD * 2);
  _Float16* dftBs  = (_Float16*)alloc((size_t)NPAD * NPAD * 2);
  _Float16* idftBs = (_Float16*)alloc((size_t)NPAD * NPAD * 2);
  _Float16* PinvBs = (_Float16*)alloc((size_t)MDIM * TDIM * LBAND * 2);
  _Float16* PsynBs = (_Float16*)alloc((size_t)MDIM * LBAND * TDIM * 2);
  _Float16* Fx     = (_Float16*)alloc((size_t)2 * MDIM * 128 * TDIM * 2);
  _Float16* Fw     = (_Float16*)alloc((size_t)2 * MDIM * 64  * TDIM * 2);
  _Float16* Xlm    = (_Float16*)alloc((size_t)2 * MDIM * 128 * LBAND * 2);
  _Float16* Wlm    = (_Float16*)alloc((size_t)2 * MDIM * 64  * LBAND * 2);
  _Float16* Olm    = (_Float16*)alloc((size_t)2 * MDIM * 128 * LBAND * 2);
  _Float16* Gm     = (_Float16*)alloc((size_t)16384 * NPAD * 2);

  // Precompute (all on stream; graph-capture safe)
  k_build_tables<<<256, 256, 0, stream>>>(dftBs, idftBs);
  k_cvt_pad   <<<16384, 256, 0, stream>>>(x,   xh, 16384);
  k_cvt_pad   <<<8192,  256, 0, stream>>>(wgt, wh, 8192);
  k_cvt_pmats <<<8192,  256, 0, stream>>>(Psyn, Pinv, PinvBs, PsynBs);

  // Pipeline
  k_dft_fwd     <<<1024, 128, 0, stream>>>(xh, dftBs, Fx, 128);
  k_dft_fwd     <<<512,  128, 0, stream>>>(wh, dftBs, Fw, 64);
  k_legendre_fwd<<<256,  256, 0, stream>>>(Fx, PinvBs, Xlm, 128);
  k_legendre_fwd<<<256,  256, 0, stream>>>(Fw, PinvBs, Wlm, 64);
  k_chanmix     <<<8192, 256, 0, stream>>>(Xlm, Wlm, Olm);
  k_legendre_inv<<<256,  256, 0, stream>>>(Olm, PsynBs, Gm);
  k_idft        <<<1024, 128, 0, stream>>>(Gm, idftBs, bias, out);
}
